// SparseAttention_29180007809107
// MI455X (gfx1250) — matrix operand order repaired and checked by reading, not compiled, not run on hardware
//
#include <hip/hip_runtime.h>

#if defined(__has_builtin)
#if __has_builtin(__builtin_amdgcn_global_load_async_to_lds_b128)
#define HAVE_ASYNC_LDS 1
#endif
#if __has_builtin(__builtin_amdgcn_s_wait_asynccnt)
#define HAVE_WAIT_ASYNC_BUILTIN 1
#endif
#endif

// ---------------------------------------------------------------------------
// Types for CDNA5 WMMA (wave32): bf16 A/B fragments (16 elems/lane), f32 accum
// ---------------------------------------------------------------------------
typedef __bf16 v16bf __attribute__((ext_vector_type(16)));
typedef __bf16 v8bf  __attribute__((ext_vector_type(8)));
typedef float  v8f   __attribute__((ext_vector_type(8)));
typedef int    v4ig  __attribute__((vector_size(16)));   // matches builtin sig

#if defined(HAVE_ASYNC_LDS)
typedef __attribute__((address_space(1))) v4ig* as1_v4i;
typedef __attribute__((address_space(3))) v4ig* as3_v4i;
#endif

static __device__ __forceinline__ __bf16 f2bf(float f) {
    union { float f; unsigned u; } in; in.f = f;
    unsigned lsb = (in.u >> 16) & 1u;
    unsigned r = in.u + 0x7fffu + lsb;         // round-to-nearest-even
    unsigned short h = (unsigned short)(r >> 16);
    union { unsigned short u; __bf16 b; } out; out.u = h;
    return out.b;
}

static __device__ __forceinline__ v16bf mk16(v8bf lo, v8bf hi) {
    v16bf r;
#pragma unroll
    for (int i = 0; i < 8; ++i) { r[i] = lo[i]; r[i + 8] = hi[i]; }
    return r;
}

static __device__ __forceinline__ v8f wmma_bf16(v16bf a, v16bf b, v8f c) {
    // v_wmma_f32_16x16x32_bf16  (D = A[16x32] * B[32x16] + C)
    return __builtin_amdgcn_wmma_f32_16x16x32_bf16(false, a, false, b,
                                                   (short)0, c, false, false);
}

// A-fragment gather: lane(m + 16h) holds k in [16h,16h+16)
static __device__ __forceinline__ v16bf load_afrag(const __bf16* row, int k0,
                                                   int h) {
    v8bf lo = *(const v8bf*)(row + k0 + 16 * h);
    v8bf hi = *(const v8bf*)(row + k0 + 16 * h + 8);
    return mk16(lo, hi);
}

static __device__ __forceinline__ void wait_async(int n) {
#if defined(HAVE_ASYNC_LDS)
#if defined(HAVE_WAIT_ASYNC_BUILTIN)
    if (n == 0)
        __builtin_amdgcn_s_wait_asynccnt(0);
    else
        __builtin_amdgcn_s_wait_asynccnt(1);
#else
    if (n == 0)
        asm volatile("s_wait_asynccnt 0x0" ::: "memory");
    else
        asm volatile("s_wait_asynccnt 0x1" ::: "memory");
#endif
#else
    (void)n;
#endif
}

// ---------------------------------------------------------------------------
// Problem constants
// ---------------------------------------------------------------------------
#define BATCH 2
#define SEQ   2048
#define HID   1024
#define NHEAD 16
#define HDIM  64
#define WIN   256
#define MROWS (BATCH * SEQ)   // 4096

// ---------------------------------------------------------------------------
// fp32 -> bf16 conversion
// ---------------------------------------------------------------------------
__global__ void k_f32_to_bf16(const float* __restrict__ in,
                              __bf16* __restrict__ out, int n) {
    int i = blockIdx.x * blockDim.x + threadIdx.x;
    if (i < n) out[i] = f2bf(in[i]);
}

// ---------------------------------------------------------------------------
// GEMM: C[M,N] = A[M,K] @ W[N,K]^T + bias[N]   (bf16 in, f32 out)
// Block = 8 waves; all waves share one 32-row A stripe (mt = blockIdx>>1),
// each wave owns a 64-col B stripe. Per K-step (32):
//   - A tile (32x32 bf16, 2 KB) staged to LDS ONCE per block via async
//     global->LDS DMA (ASYNCcnt), double-buffered so the DMA for step k+1
//     overlaps the 8 WMMAs of step k.  (Falls back to ds_store staging.)
//   - B fragments are wave-unique, register double-buffered as before.
// ---------------------------------------------------------------------------
__global__ __launch_bounds__(256)
void k_gemm_bf16(const __bf16* __restrict__ A, const __bf16* __restrict__ W,
                 const float* __restrict__ bias, float* __restrict__ C,
                 int M, int N, int K) {
    __shared__ __attribute__((aligned(64))) __bf16 asmem[2][32][32];

    const int t    = threadIdx.x;
    const int lane = t & 31;
    const int wv   = t >> 5;                    // wave in block, 0..7
    const int mt   = blockIdx.x >> 1;           // 32-row tile (block-uniform)
    const int nt   = ((blockIdx.x & 1) << 3) + wv;  // 64-col tile per wave
    if (mt * 32 >= M) return;                   // uniform

    const int n = lane & 15;
    const int h = lane >> 4;

    // ---- block-wide A-tile stager: 128 lanes x 16B = 32x32 bf16 ----
    const int srow = t >> 2, schk = t & 3;      // row 0..31, 16B chunk 0..3
    const __bf16* sgp = A + (size_t)(mt * 32 + srow) * K + schk * 8;
    __bf16* slp = &asmem[0][srow][schk * 8];
    auto stage = [&](int buf, int k0) {
        if (t < 128) {
#if defined(HAVE_ASYNC_LDS)
            __builtin_amdgcn_global_load_async_to_lds_b128(
                (as1_v4i)(sgp + k0),
                (as3_v4i)(slp + buf * (32 * 32)),
                0, 0);
#else
            *(v8bf*)(slp + buf * (32 * 32)) = *(const v8bf*)(sgp + k0);
#endif
        }
    };

    const __bf16* wrow[4];
#pragma unroll
    for (int j = 0; j < 4; ++j)
        wrow[j] = W + (size_t)(nt * 64 + j * 16 + n) * K;

    v8f zero = {};
    v8f acc[2][4] = {{zero, zero, zero, zero}, {zero, zero, zero, zero}};

    // ---- prologue: stage A tile 0, load B fragments for step 0 ----
    stage(0, 0);
    v16bf bCur[4], bNxt[4];
#pragma unroll
    for (int j = 0; j < 4; ++j)
        bCur[j] = *(const v16bf*)(wrow[j] + 16 * h);

    const int kSteps = K >> 5;
    int buf = 0;
    for (int ks = 0; ks < kSteps; ++ks) {
        const int pending = (ks + 1 < kSteps) ? 1 : 0;
        const int kn = pending ? ((ks + 1) << 5) : 0;
        if (pending) stage(buf ^ 1, kn);        // async DMA for next tile
#pragma unroll
        for (int j = 0; j < 4; ++j)             // next B frags (dummy on last)
            bNxt[j] = *(const v16bf*)(wrow[j] + kn + 16 * h);

        wait_async(pending);                    // this tile's DMA done
        __syncthreads();                        // ... for every wave

        // A fragments from LDS (rows m*16+n of the 32x32 tile)
        v16bf aCur[2];
#pragma unroll
        for (int m = 0; m < 2; ++m) {
            const __bf16* ar = &asmem[buf][m * 16 + n][0];
            aCur[m] = mk16(*(const v8bf*)(ar + 16 * h),
                           *(const v8bf*)(ar + 16 * h + 8));
        }

        // 8 independent WMMAs; B regs are one step old (loads overlap)
#pragma unroll
        for (int m = 0; m < 2; ++m)
#pragma unroll
            for (int j = 0; j < 4; ++j)
                acc[m][j] = wmma_bf16(aCur[m], bCur[j], acc[m][j]);

#pragma unroll
        for (int j = 0; j < 4; ++j) bCur[j] = bNxt[j];

        __syncthreads();                        // WAR: buffer re-staged next it
        buf ^= 1;
    }

    // C/D layout: lane l, vgpr g -> row (g + 8h), col (l&15)
#pragma unroll
    for (int m = 0; m < 2; ++m) {
#pragma unroll
        for (int j = 0; j < 4; ++j) {
#pragma unroll
            for (int g = 0; g < 8; ++g) {
                int row = mt * 32 + m * 16 + g + 8 * h;
                int col = nt * 64 + j * 16 + n;
                C[(size_t)row * N + col] = acc[m][j][g] + bias[col];
            }
        }
    }
}

// ---------------------------------------------------------------------------
// RoPE + repack: qlin/klin/vlin are row-major (B*L, 1024) f32.
// Emits q,k as (B,H,L,D) bf16 (q pre-scaled by 1/sqrt(D)) and v transposed
// to (B,H,D,L) bf16 so attention B-fragments are contiguous loads.
// ---------------------------------------------------------------------------
__global__ void k_rope_pack(const float* __restrict__ qlin,
                            const float* __restrict__ klin,
                            const float* __restrict__ vlin,
                            __bf16* __restrict__ qbf,
                            __bf16* __restrict__ kbf,
                            __bf16* __restrict__ vtbf) {
    int idx = blockIdx.x * blockDim.x + threadIdx.x;
    const int total = BATCH * NHEAD * SEQ * HDIM;
    if (idx >= total) return;
    int d  = idx & 63;
    int i  = (idx >> 6) & (SEQ - 1);
    int hh = (idx >> 17) & 15;
    int b  = idx >> 21;

    size_t lin = ((size_t)(b * SEQ + i)) * HID + hh * HDIM + d;
    float xq = qlin[lin], xk = klin[lin], xv = vlin[lin];
    size_t lin2 = (d < 32) ? (lin + 32) : (lin - 32);
    float sgn = (d < 32) ? -1.0f : 1.0f;
    float rq = sgn * qlin[lin2];
    float rk = sgn * klin[lin2];

    // inv_freq = 10000^(-(d%32)/32) = exp(-ln(10000)/32 * (d%32))
    float invf = __expf(-0.28782313662425575f * (float)(d & 31));
    float ang = (float)i * invf;
    float s, c;
    __sincosf(ang, &s, &c);

    size_t qo = (((size_t)(b * NHEAD + hh)) * SEQ + i) * HDIM + d;
    qbf[qo] = f2bf((xq * c + rq * s) * 0.125f);   // fold 1/sqrt(64)
    kbf[qo] = f2bf(xk * c + rk * s);
    size_t vo = (((size_t)(b * NHEAD + hh)) * HDIM + d) * SEQ + i;
    vtbf[vo] = f2bf(xv);
}

// ---------------------------------------------------------------------------
// Banded flash attention: one wave per (b,h, 16-query tile).
// Per iteration: 32 keys (two 16-key tiles) -> 4 WMMA for Q@K^T (K-dim=64)
// + online softmax + 4 WMMA for P@V (K=32, d=64). P goes C-layout -> LDS ->
// A-layout (bf16). Window mask: j<=i && (i-j<WIN || j==0).
// ---------------------------------------------------------------------------
__global__ __launch_bounds__(128)
void k_attn(const __bf16* __restrict__ qbf, const __bf16* __restrict__ kbf,
            const __bf16* __restrict__ vtbf, __bf16* __restrict__ attn) {
    __shared__ __attribute__((aligned(64))) __bf16 pl[4][16][32];

    const int lane = threadIdx.x & 31;
    const int wl   = threadIdx.x >> 5;
    const int wave = blockIdx.x * 4 + wl;
    const int qt = wave & (SEQ / 16 - 1);       // 0..127
    const int bh = wave >> 7;                   // 0..31
    if (bh >= BATCH * NHEAD) return;            // wave-uniform
    const int n = lane & 15;
    const int h = lane >> 4;

    const __bf16* qb = qbf  + (size_t)bh * (SEQ * HDIM);
    const __bf16* kb = kbf  + (size_t)bh * (SEQ * HDIM);
    const __bf16* vb = vtbf + (size_t)bh * (HDIM * SEQ);

    // Q A-fragments (row = query qt*16 + n), K-dim 64 -> 2 fragments
    const __bf16* qrow = qb + (size_t)(qt * 16 + n) * HDIM;
    v16bf qf[2];
#pragma unroll
    for (int s2 = 0; s2 < 2; ++s2)
        qf[s2] = load_afrag(qrow, 32 * s2, h);

    float rm[8], rl[8];
    v8f zero = {};
    v8f acc[4] = {zero, zero, zero, zero};
#pragma unroll
    for (int g = 0; g < 8; ++g) { rm[g] = -1e30f; rl[g] = 0.0f; }

    // key-pair tiles covering the sliding window, plus the global tile (kp=0)
    int kpStart = (qt > 16) ? ((qt - 16) >> 1) : 0;
    const int kpEnd = qt >> 1;
    const int extra = (kpStart > 0) ? 1 : 0;
    const int nIter = kpEnd - kpStart + 1 + extra;

    for (int it = 0; it < nIter; ++it) {
        int kp = extra ? ((it == 0) ? 0 : (kpStart + it - 1)) : (kpStart + it);

        // ---- S = Q @ K^T (scores for 32 keys, pre-scaled via q) ----
        v8f s0 = zero, s1 = zero;
        const __bf16* kr0 = kb + (size_t)(kp * 32 + n) * HDIM;
        const __bf16* kr1 = kb + (size_t)(kp * 32 + 16 + n) * HDIM;
#pragma unroll
        for (int s2 = 0; s2 < 2; ++s2) {
            v16bf b0 = *(const v16bf*)(kr0 + 32 * s2 + 16 * h);
            v16bf b1 = *(const v16bf*)(kr1 + 32 * s2 + 16 * h);
            s0 = wmma_bf16(qf[s2], b0, s0);
            s1 = wmma_bf16(qf[s2], b1, s1);
        }

        // ---- mask + online softmax (row = g + 8h, cols = n / n+16) ----
        float alpha[8];
#pragma unroll
        for (int g = 0; g < 8; ++g) {
            int irow = qt * 16 + g + 8 * h;
            int j0 = kp * 32 + n;
            int j1 = j0 + 16;
            float v0 = s0[g];
            float v1 = s1[g];
            if (j0 > irow || ((irow - j0 >= WIN) && j0 != 0)) v0 = -1e30f;
            if (j1 > irow || ((irow - j1 >= WIN) && j1 != 0)) v1 = -1e30f;
            float mx = fmaxf(v0, v1);
#pragma unroll
            for (int off = 1; off < 16; off <<= 1)
                mx = fmaxf(mx, __shfl_xor(mx, off, 32));
            float mnew = fmaxf(rm[g], mx);
            alpha[g] = __expf(rm[g] - mnew);
            float p0 = (v0 < -1e29f) ? 0.0f : __expf(v0 - mnew);
            float p1 = (v1 < -1e29f) ? 0.0f : __expf(v1 - mnew);
            float sm = p0 + p1;
#pragma unroll
            for (int off = 1; off < 16; off <<= 1)
                sm += __shfl_xor(sm, off, 32);
            rl[g] = rl[g] * alpha[g] + sm;
            rm[g] = mnew;
            pl[wl][g + 8 * h][n]      = f2bf(p0);
            pl[wl][g + 8 * h][n + 16] = f2bf(p1);
        }

        // rescale accumulators
#pragma unroll
        for (int dt = 0; dt < 4; ++dt)
#pragma unroll
            for (int g = 0; g < 8; ++g) acc[dt][g] *= alpha[g];

        // LDS RAW: DS ops are wave-ordered; wait + stop compiler reordering
        asm volatile("s_wait_dscnt 0" ::: "memory");

        // P as A-fragment (row = n, k chunks at 16h and 16h+8)
        const __bf16* prow = &pl[wl][n][0];
        v16bf pa = mk16(*(const v8bf*)(prow + 16 * h),
                        *(const v8bf*)(prow + 16 * h + 8));

        // ---- acc += P @ V  (V^T rows are d, contiguous along keys) ----
#pragma unroll
        for (int dt = 0; dt < 4; ++dt) {
            const __bf16* vrow = vb + (size_t)(dt * 16 + n) * SEQ;
            v16bf vfr = *(const v16bf*)(vrow + kp * 32 + 16 * h);
            acc[dt] = wmma_bf16(pa, vfr, acc[dt]);
        }
        asm volatile("s_wait_dscnt 0" ::: "memory");
    }

    // epilogue: out = acc / l, written row-major (B*L, 1024) as bf16
    const int b = bh >> 4;
    const int head = bh & 15;
#pragma unroll
    for (int dt = 0; dt < 4; ++dt) {
#pragma unroll
        for (int g = 0; g < 8; ++g) {
            int irow = qt * 16 + g + 8 * h;
            float o = acc[dt][g] / rl[g];
            attn[(size_t)(b * SEQ + irow) * HID + head * HDIM + dt * 16 + n] =
                f2bf(o);
        }
    }
}

// ---------------------------------------------------------------------------
// Host launcher
// ---------------------------------------------------------------------------
extern "C" void kernel_launch(void* const* d_in, const int* in_sizes, int n_in,
                              void* d_out, int out_size, void* d_ws,
                              size_t ws_size, hipStream_t stream) {
    const float* hs  = (const float*)d_in[0];
    const float* q_w = (const float*)d_in[1];
    const float* q_b = (const float*)d_in[2];
    const float* k_w = (const float*)d_in[3];
    const float* k_b = (const float*)d_in[4];
    const float* v_w = (const float*)d_in[5];
    const float* v_b = (const float*)d_in[6];
    const float* o_w = (const float*)d_in[7];
    const float* o_b = (const float*)d_in[8];
    float* out = (float*)d_out;

    const int M = MROWS, N = HID, K = HID;
    char* ws = (char*)d_ws;
    size_t off = 0;
    auto carve = [&](size_t bytes) -> void* {
        void* p = ws + off;
        off += (bytes + 255) & ~(size_t)255;
        return p;
    };

    __bf16* hs_bf = (__bf16*)carve((size_t)M * K * 2);
    __bf16* qw_bf = (__bf16*)carve((size_t)N * K * 2);
    __bf16* kw_bf = (__bf16*)carve((size_t)N * K * 2);
    __bf16* vw_bf = (__bf16*)carve((size_t)N * K * 2);
    __bf16* ow_bf = (__bf16*)carve((size_t)N * K * 2);
    float* qlin = (float*)carve((size_t)M * N * 4);
    float* klin = (float*)carve((size_t)M * N * 4);
    float* vlin = (float*)carve((size_t)M * N * 4);
    __bf16* q_bf  = (__bf16*)carve((size_t)M * N * 2);
    __bf16* k_bf  = (__bf16*)carve((size_t)M * N * 2);
    __bf16* vt_bf = (__bf16*)carve((size_t)M * N * 2);
    __bf16* attn_bf = (__bf16*)qlin;   // qlin dead after RoPE; reuse

    // 1) fp32 -> bf16 conversions
    {
        int n = M * K;
        k_f32_to_bf16<<<(n + 255) / 256, 256, 0, stream>>>(hs, hs_bf, n);
        n = N * K;
        k_f32_to_bf16<<<(n + 255) / 256, 256, 0, stream>>>(q_w, qw_bf, n);
        k_f32_to_bf16<<<(n + 255) / 256, 256, 0, stream>>>(k_w, kw_bf, n);
        k_f32_to_bf16<<<(n + 255) / 256, 256, 0, stream>>>(v_w, vw_bf, n);
        k_f32_to_bf16<<<(n + 255) / 256, 256, 0, stream>>>(o_w, ow_bf, n);
    }

    // 2) QKV projections: blocks = (M/32)*2, block-uniform A stripe
    const int gemmBlocks = (M / 32) * (N / 64) / 8;
    k_gemm_bf16<<<gemmBlocks, 256, 0, stream>>>(hs_bf, qw_bf, q_b, qlin, M, N, K);
    k_gemm_bf16<<<gemmBlocks, 256, 0, stream>>>(hs_bf, kw_bf, k_b, klin, M, N, K);
    k_gemm_bf16<<<gemmBlocks, 256, 0, stream>>>(hs_bf, vw_bf, v_b, vlin, M, N, K);

    // 3) RoPE + repack to attention-friendly bf16 layouts
    {
        int n = BATCH * NHEAD * SEQ * HDIM;
        k_rope_pack<<<(n + 255) / 256, 256, 0, stream>>>(qlin, klin, vlin,
                                                         q_bf, k_bf, vt_bf);
    }

    // 4) banded attention: B*H*(L/16) = 4096 wave tiles, 4 waves/block
    k_attn<<<BATCH * NHEAD * (SEQ / 16) / 4, 128, 0, stream>>>(q_bf, k_bf,
                                                               vt_bf, attn_bf);

    // 5) output projection -> f32 d_out
    k_gemm_bf16<<<gemmBlocks, 256, 0, stream>>>(attn_bf, ow_bf, o_b, out, M, N, K);
}
